// MultiHeadLatentAttention_68186900791573
// MI455X (gfx1250) — matrix operand order repaired and checked by reading, not compiled, not run on hardware
//
#include <hip/hip_runtime.h>
#include <hip/hip_bf16.h>
#include <math.h>

// ---------------- CDNA5 WMMA plumbing (wave32) ----------------
typedef __bf16 bf16_t;
typedef __attribute__((ext_vector_type(16))) __bf16 v16bf;
typedef __attribute__((ext_vector_type(8)))  float  v8f;

#define WMMA_BF16(a, b, c) \
  __builtin_amdgcn_wmma_f32_16x16x32_bf16(false, (a), false, (b), (short)0, (c), false, false)

// ---------------- optional Tensor Data Mover path ----------------
#if defined(__has_builtin)
#if __has_builtin(__builtin_amdgcn_tensor_load_to_lds) && __has_builtin(__builtin_amdgcn_s_wait_tensorcnt)
#define HAVE_TDM 1
#endif
#endif

#ifdef HAVE_TDM
typedef __attribute__((ext_vector_type(4))) unsigned int tdm_v4u;
typedef __attribute__((ext_vector_type(4))) int          tdm_v4i;
typedef __attribute__((ext_vector_type(8))) int          tdm_v8i;

// 1-D contiguous global->LDS DMA of 2-byte elements via the TDM (D# per ISA 8.3/8.4).
__device__ inline void tdm_load_1d(const void* gsrc, unsigned lds_off, unsigned nbytes) {
  const unsigned nelem = nbytes >> 1;                 // data_size = 2 bytes
  const unsigned long long ga = (unsigned long long)gsrc;
  tdm_v4u g0 = {};
  g0.x = 1u;                                          // count=1, user descriptor
  g0.y = lds_off;                                     // lds_addr (bytes)
  g0.z = (unsigned)ga;                                // global_addr[31:0]
  g0.w = (unsigned)(ga >> 32) | (2u << 30);           // global_addr[56:32] | type=2
  tdm_v8i g1 = {};
  g1[0] = (int)(1u << 16);                            // data_size=1 (2B), wg_mask=0
  g1[1] = (int)((nelem & 0xffffu) << 16);             // tensor_dim0[15:0]
  g1[2] = (int)(((nelem >> 16) & 0xffffu) | (1u << 16)); // tensor_dim0[31:16] | tensor_dim1=1
  g1[3] = (int)((nelem & 0xffffu) << 16);             // tile_dim0
  g1[5] = (int)nelem;                                 // tensor_dim0_stride (lo)
  tdm_v4i z4 = {};
#if __clang_major__ >= 23
  tdm_v8i z8 = {};
  __builtin_amdgcn_tensor_load_to_lds(g0, g1, z4, z4, z8, 0);
#else
  __builtin_amdgcn_tensor_load_to_lds(g0, g1, z4, z4, 0);
#endif
}
#endif

// ---------------- problem constants ----------------
#define EMBED   1536
#define NHEADS  16
#define DHEAD   96
#define QPROJ   768
#define KVPROJ  1024
#define CKV_W   1072   // KV_PROJ + ROPE
#define KV_W    2304   // EMBED + NUM_HEADS*NOPE
#define NOPE    48
#define ROPE    48
#define SEQ     1024
#define BATCH   4
#define MTOT    (BATCH*SEQ)   // 4096

// ---- fragment loaders (per ISA 7.12.2 16-bit layouts, wave32) ----
// A 16x32 bf16 from LDS row-major [M][K]: per lane = 2 contiguous 16B runs -> ds_load_b128 x2
__device__ inline v16bf load_frag_a(const bf16_t* __restrict__ base, int lane, int ld) {
  const int m = lane & 15;
  const int h = lane >> 4;
  v16bf f;
#pragma unroll
  for (int vv = 0; vv < 8; ++vv) {
    const int kb = (vv < 4) ? (8*h + 2*vv) : (16 + 8*h + 2*(vv - 4));
    f[2*vv]     = base[m*ld + kb];
    f[2*vv + 1] = base[m*ld + kb + 1];
  }
  return f;
}
// B 32x16 bf16 from LDS stored K-contiguous per column ([N][K], stride ld):
// per lane = 2 contiguous 16B runs (k = 8h..8h+7, 16+8h..16+8h+7) -> ds_load_b128 x2
__device__ inline v16bf load_frag_bt(const bf16_t* __restrict__ base, int lane, int ld) {
  const int n  = lane & 15;
  const int kb = (lane >> 4) * 8;
  v16bf f;
#pragma unroll
  for (int e = 0; e < 16; ++e) f[e] = base[n*ld + ((e < 8) ? (kb + e) : (16 + kb + (e - 8)))];
  return f;
}

// ---------------- tiled bf16 WMMA GEMM: C(MxN) = A(MxK) * B ----------------
// TRANSB=false: B is KxN row-major.  TRANSB=true: B is NxK row-major (C = A * B^T).
// Block tile 128x128, 8 waves, wave tile 32x64 (8 WMMA / K-step).
// M % 128 == 0, K % 32 == 0 required; N edge handled with guards.
template <bool TRANSB>
__global__ __launch_bounds__(256)
void gemm_bf16(const float* __restrict__ A, const float* __restrict__ B,
               float* __restrict__ C, int M, int N, int K) {
  __shared__ bf16_t As[128 * 32];    // [m][k]
  __shared__ bf16_t Bst[128 * 32];   // [n][k]  (K-contiguous, fragment-friendly)
  const int tid  = threadIdx.x;
  const int lane = tid & 31;
  const int wid  = tid >> 5;
  const int wm   = wid & 3;          // 4 row groups of 32
  const int wn   = wid >> 2;         // 2 col groups of 64
  const int bm   = blockIdx.y * 128;
  const int bn   = blockIdx.x * 128;

  v8f acc[2][4] = {};
  for (int k0 = 0; k0 < K; k0 += 32) {
    if (k0 + 32 < K)  // gfx1250 global_prefetch of next A tile
      __builtin_prefetch(&A[(size_t)(bm + (tid >> 1)) * K + k0 + 32], 0, 1);
    // A tile 128x32: float4 loads, convert to bf16
#pragma unroll
    for (int i = 0; i < 4; ++i) {
      const int f = i*256 + tid;                  // float4 index
      const int r = f >> 3, c = (f & 7) * 4;
      const float4 av = *(const float4*)&A[(size_t)(bm + r)*K + k0 + c];
      bf16_t* dst = &As[r*32 + c];
      dst[0] = (bf16_t)av.x; dst[1] = (bf16_t)av.y;
      dst[2] = (bf16_t)av.z; dst[3] = (bf16_t)av.w;
    }
    // B tile -> LDS transposed [n][k] (zero-pad N edge)
#pragma unroll
    for (int i = 0; i < 16; ++i) {
      const int idx = i*256 + tid;
      const int n = idx >> 5, kk = idx & 31;
      const int col = bn + n;
      float vl = 0.f;
      if (col < N)
        vl = TRANSB ? B[(size_t)col*K + k0 + kk] : B[(size_t)(k0 + kk)*N + col];
      Bst[idx] = (bf16_t)vl;
    }
    __syncthreads();
    v16bf af[2], bfr[4];
#pragma unroll
    for (int i = 0; i < 2; ++i) af[i]  = load_frag_a(&As[(wm*32 + i*16)*32], lane, 32);
#pragma unroll
    for (int j = 0; j < 4; ++j) bfr[j] = load_frag_bt(&Bst[(wn*64 + j*16)*32], lane, 32);
#pragma unroll
    for (int i = 0; i < 2; ++i)
#pragma unroll
      for (int j = 0; j < 4; ++j)
        acc[i][j] = WMMA_BF16(af[i], bfr[j], acc[i][j]);
    __syncthreads();
  }
  const int half = lane >> 4, n16 = lane & 15;
#pragma unroll
  for (int i = 0; i < 2; ++i)
#pragma unroll
    for (int j = 0; j < 4; ++j) {
      const int row0 = bm + wm*32 + i*16 + half*8;
      const int col  = bn + wn*64 + j*16 + n16;
      if (col < N)
#pragma unroll
        for (int r = 0; r < 8; ++r)
          C[(size_t)(row0 + r)*N + col] = acc[i][j][r];
    }
}

// ---------------- row LayerNorm (block per row) ----------------
__global__ __launch_bounds__(256)
void layernorm_rows(const float* __restrict__ X, float* __restrict__ Y,
                    const float* __restrict__ g, const float* __restrict__ b,
                    int W, int strideX, int strideY) {
  __shared__ float red[256];
  const int row = blockIdx.x, tid = threadIdx.x;
  const float* x = X + (size_t)row * strideX;
  float s = 0.f;
  for (int j = tid; j < W; j += 256) s += x[j];
  red[tid] = s; __syncthreads();
  for (int o = 128; o > 0; o >>= 1) { if (tid < o) red[tid] += red[tid + o]; __syncthreads(); }
  const float mean = red[0] / W; __syncthreads();
  float v = 0.f;
  for (int j = tid; j < W; j += 256) { const float d = x[j] - mean; v += d*d; }
  red[tid] = v; __syncthreads();
  for (int o = 128; o > 0; o >>= 1) { if (tid < o) red[tid] += red[tid + o]; __syncthreads(); }
  const float rstd = rsqrtf(red[0] / W + 1e-5f); __syncthreads();
  float* y = Y + (size_t)row * strideY;
  for (int j = tid; j < W; j += 256) y[j] = (x[j] - mean) * rstd * g[j] + b[j];
}

// ---------------- RoPE + head split into bf16 head-major buffers ----------------
__global__ __launch_bounds__(256)
void build_heads(const float* __restrict__ Q, const float* __restrict__ KV,
                 const float* __restrict__ CKV, const float* __restrict__ cosc,
                 const float* __restrict__ sinc,
                 bf16_t* __restrict__ qh, bf16_t* __restrict__ kh, bf16_t* __restrict__ vh) {
  const int bs = blockIdx.x;          // b*SEQ + s
  const int s  = bs & (SEQ - 1);
  const int tid = threadIdx.x;
  __shared__ float kr[ROPE];          // roped K_rope, shared across heads
  if (tid < ROPE) {
    const int dd = tid;
    const int f  = (dd < 24) ? dd : dd - 24;
    const float c  = cosc[s*48 + f], sn = sinc[s*48 + f];
    const float x  = CKV[(size_t)bs*CKV_W + KVPROJ + dd];
    const float xo = (dd < 24) ? -CKV[(size_t)bs*CKV_W + KVPROJ + dd + 24]
                               :  CKV[(size_t)bs*CKV_W + KVPROJ + dd - 24];
    kr[dd] = x*c + xo*sn;
  }
  __syncthreads();
  const size_t qrow  = (size_t)bs * EMBED;
  const size_t kvrow = (size_t)bs * KV_W;
  for (int i = tid; i < NHEADS*DHEAD; i += 256) {
    const int h = i / DHEAD, d = i % DHEAD;
    const size_t o = ((size_t)((bs >> 10)*NHEADS + h) * SEQ + s) * DHEAD + d;
    float qv;
    if (d < NOPE) qv = Q[qrow + h*DHEAD + d];
    else {
      const int dd = d - NOPE;
      const int f  = (dd < 24) ? dd : dd - 24;
      const float c  = cosc[s*48 + f], sn = sinc[s*48 + f];
      const float x  = Q[qrow + h*DHEAD + NOPE + dd];
      const float xo = (dd < 24) ? -Q[qrow + h*DHEAD + NOPE + dd + 24]
                                 :  Q[qrow + h*DHEAD + NOPE + dd - 24];
      qv = x*c + xo*sn;
    }
    qh[o] = (bf16_t)qv;
    kh[o] = (bf16_t)((d < NOPE) ? KV[kvrow + h*(DHEAD + NOPE) + d] : kr[d - NOPE]);
    vh[o] = (bf16_t)KV[kvrow + h*(DHEAD + NOPE) + NOPE + d];
  }
}

// ---------------- fused causal attention, 16 queries per block ----------------
// dynamic LDS layout (byte offsets):
#define SM_SCORES 0                         // f32 [16][1024]            65536
#define SM_P      65536                     // bf16[16][1024]            32768
#define SM_QT     (65536 + 32768)           // bf16[16][96]               3072
#define SM_KT     (SM_QT + 3072)            // bf16[128][96]             24576
#define SM_VT     (SM_KT + 24576)           // bf16[96][32] (transposed)  6144
#define SM_RED    (SM_VT + 6144)            // f32 [16][16]               1024
#define SM_STAT   (SM_RED + 1024)           // f32 [16]                    128
#define SM_TOTAL  (SM_STAT + 128)
__global__ __launch_bounds__(256)
void mla_attention(const bf16_t* __restrict__ qh, const bf16_t* __restrict__ kh,
                   const bf16_t* __restrict__ vh, float* __restrict__ AO) {
  extern __shared__ char smem[];
  float (*scores)[SEQ] = (float (*)[SEQ])(smem + SM_SCORES);
  bf16_t (*P)[SEQ]     = (bf16_t (*)[SEQ])(smem + SM_P);
  bf16_t* qt           = (bf16_t*)(smem + SM_QT);
  bf16_t* kt           = (bf16_t*)(smem + SM_KT);
  bf16_t* vtT          = (bf16_t*)(smem + SM_VT);   // [d][key] d-major
  float (*red)[16]     = (float (*)[16])(smem + SM_RED);
  float* rowstat       = (float*)(smem + SM_STAT);

  const int tid = threadIdx.x, lane = tid & 31, wid = tid >> 5;
  const int qbase = blockIdx.x * 16;
  const int h = blockIdx.y, b = blockIdx.z;
  const size_t head_off = (size_t)(b*NHEADS + h) * SEQ * DHEAD;
  const bf16_t* q = qh + head_off + (size_t)qbase * DHEAD;
  const bf16_t* k = kh + head_off;
  const bf16_t* v = vh + head_off;

  for (int i = tid; i < 16*DHEAD; i += 256) qt[i] = q[i];
  __syncthreads();

  const float scale = 0.10206207261596575f;   // 1/sqrt(96)
  const int half = lane >> 4, n16 = lane & 15;

  // ---- scores = scale * Q K^T, causal masked ----
  for (int kc = 0; kc < SEQ; kc += 128) {
#ifdef HAVE_TDM
    if (wid == 0) {           // TDM DMA: 128x96 bf16 K tile, contiguous both sides
      tdm_load_1d(k + (size_t)kc*DHEAD, SM_KT, 128*DHEAD*2);
      __builtin_amdgcn_s_wait_tensorcnt(0);
    }
#else
    for (int i = tid; i < 128*DHEAD; i += 256) kt[i] = k[(size_t)kc*DHEAD + i];
#endif
    __syncthreads();
    v8f sacc = {};
#pragma unroll
    for (int ks = 0; ks < DHEAD; ks += 32) {
      const v16bf a  = load_frag_a(qt + ks, lane, DHEAD);
      const v16bf bb = load_frag_bt(kt + wid*16*DHEAD + ks, lane, DHEAD);  // K^T
      sacc = WMMA_BF16(a, bb, sacc);
    }
    const int key = kc + wid*16 + n16;
#pragma unroll
    for (int r = 0; r < 8; ++r) {
      const int qr = r + 8*half;
      scores[qr][key] = (key <= qbase + qr) ? sacc[r]*scale : -1e30f;
    }
    __syncthreads();
  }

  // ---- softmax over the 16 rows (16 threads per row) ----
  {
    const int row = tid >> 4, c16 = tid & 15;
    float m = -1e30f;
    for (int j = c16; j < SEQ; j += 16) m = fmaxf(m, scores[row][j]);
    red[row][c16] = m; __syncthreads();
    if (c16 == 0) { float mm = -1e30f; for (int t = 0; t < 16; ++t) mm = fmaxf(mm, red[row][t]); rowstat[row] = mm; }
    __syncthreads();
    const float rm = rowstat[row];
    float sum = 0.f;
    for (int j = c16; j < SEQ; j += 16) { const float e = __expf(scores[row][j] - rm); scores[row][j] = e; sum += e; }
    red[row][c16] = sum; __syncthreads();
    if (c16 == 0) { float ss = 0.f; for (int t = 0; t < 16; ++t) ss += red[row][t]; rowstat[row] = ss; }
    __syncthreads();
    const float inv = 1.f / rowstat[row];
    for (int j = c16; j < SEQ; j += 16) P[row][j] = (bf16_t)(scores[row][j] * inv);
  }
  __syncthreads();

  // ---- O = P @ V : waves 0..5 each own one 16-wide d tile ----
  v8f oacc = {};
  for (int kc = 0; kc < SEQ; kc += 32) {
    for (int i = tid; i < 32*DHEAD; i += 256) {   // stage V transposed [d][key]
      const int key = i / DHEAD, d = i % DHEAD;
      vtT[d*32 + key] = v[(size_t)(kc + key)*DHEAD + d];
    }
    __syncthreads();
    if (wid < 6) {
      const v16bf a  = load_frag_a(&P[0][0] + kc, lane, SEQ);
      const v16bf bb = load_frag_bt(vtT + (wid*16)*32, lane, 32);
      oacc = WMMA_BF16(a, bb, oacc);
    }
    __syncthreads();
  }
  if (wid < 6) {
    const int row0 = qbase + 8*half;
    const int col  = h*DHEAD + wid*16 + n16;
#pragma unroll
    for (int r = 0; r < 8; ++r)
      AO[((size_t)b*SEQ + row0 + r)*EMBED + col] = oacc[r];
  }
}

// ---------------- launch ----------------
extern "C" void kernel_launch(void* const* d_in, const int* in_sizes, int n_in,
                              void* d_out, int out_size, void* d_ws, size_t ws_size,
                              hipStream_t stream) {
  const float* X    = (const float*)d_in[0];
  const float* Wdq  = (const float*)d_in[1];
  const float* Wuq  = (const float*)d_in[2];
  const float* qg   = (const float*)d_in[3];
  const float* qb   = (const float*)d_in[4];
  const float* Wdkv = (const float*)d_in[5];
  const float* Wukv = (const float*)d_in[6];
  const float* kvg  = (const float*)d_in[7];
  const float* kvb  = (const float*)d_in[8];
  const float* Wo   = (const float*)d_in[9];
  const float* cosc = (const float*)d_in[10];
  const float* sinc = (const float*)d_in[11];

  float* out = (float*)d_out;                       // (4096,1536)
  float* ckv = out + (size_t)MTOT * EMBED;          // (4096,1072) second output

  float*  cq   = (float*)d_ws;                      // (4096,768)
  float*  Qb   = cq   + (size_t)MTOT * QPROJ;       // (4096,1536)
  float*  kvln = Qb   + (size_t)MTOT * EMBED;       // (4096,1024)
  float*  KVb  = kvln + (size_t)MTOT * KVPROJ;      // (4096,2304)
  const size_t HSZ = (size_t)BATCH * NHEADS * SEQ * DHEAD;
  bf16_t* qh = (bf16_t*)(KVb + (size_t)MTOT * KV_W);
  bf16_t* kh = qh + HSZ;
  bf16_t* vh = kh + HSZ;
  float*  AO = (float*)(vh + HSZ);                  // (4096,1536)

  const dim3 blk(256);
  // 1. cq = X @ W_dq
  gemm_bf16<false><<<dim3((QPROJ + 127)/128, MTOT/128), blk, 0, stream>>>(X, Wdq, cq, MTOT, QPROJ, EMBED);
  // 2. q layernorm (in place)
  layernorm_rows<<<MTOT, blk, 0, stream>>>(cq, cq, qg, qb, QPROJ, QPROJ, QPROJ);
  // 3. Q = cq @ W_uq
  gemm_bf16<false><<<dim3((EMBED + 127)/128, MTOT/128), blk, 0, stream>>>(cq, Wuq, Qb, MTOT, EMBED, QPROJ);
  // 4. compressed_kv = X @ W_dkv  (second harness output)
  gemm_bf16<false><<<dim3((CKV_W + 127)/128, MTOT/128), blk, 0, stream>>>(X, Wdkv, ckv, MTOT, CKV_W, EMBED);
  // 5. kv layernorm over first 1024 cols
  layernorm_rows<<<MTOT, blk, 0, stream>>>(ckv, kvln, kvg, kvb, KVPROJ, CKV_W, KVPROJ);
  // 6. KV = kvln @ W_ukv
  gemm_bf16<false><<<dim3((KV_W + 127)/128, MTOT/128), blk, 0, stream>>>(kvln, Wukv, KVb, MTOT, KV_W, KVPROJ);
  // 7. RoPE + head split -> bf16
  build_heads<<<MTOT, blk, 0, stream>>>(Qb, KVb, ckv, cosc, sinc, qh, kh, vh);
  // 8. fused causal attention
  mla_attention<<<dim3(SEQ/16, NHEADS, BATCH), blk, SM_TOTAL, stream>>>(qh, kh, vh, AO);
  // 9. out = AO @ W_o^T
  gemm_bf16<true><<<dim3((EMBED + 127)/128, MTOT/128), blk, 0, stream>>>(AO, Wo, out, MTOT, EMBED, EMBED);
}
